// DWAC_72705206387155
// MI455X (gfx1250) — matrix operand order repaired and checked by reading, not compiled, not run on hardware
//
#include <hip/hip_runtime.h>
#include <hip/hip_bf16.h>

typedef __attribute__((ext_vector_type(16))) __bf16 v16bf;
typedef __attribute__((ext_vector_type(8)))  __bf16 v8bf;
typedef __attribute__((ext_vector_type(4)))  __bf16 v4bf;
typedef __attribute__((ext_vector_type(8)))  float  v8f;
typedef __attribute__((ext_vector_type(4)))  unsigned int u32x4;
typedef __attribute__((ext_vector_type(8)))  int          i32x8;
typedef __attribute__((ext_vector_type(4)))  int          i32x4;

#define NROWS 8192
#define XDIM  1024
#define DH1   512
#define DH2   256
#define NCLS  10
#define GAMMA 1.0f
#define KEPS  1e-6f
#define NJC   16
#define JCHUNK (NROWS / NJC)   // 512

#if __has_builtin(__builtin_amdgcn_tensor_load_to_lds)
#define HAVE_TDM 1
#else
#define HAVE_TDM 0
#endif

// ---------------------------------------------------------------------------
// Pre-pass: f32 -> bf16 copy (x) and f32 -> bf16 transpose (weights), so the
// GEMM tiles are raw byte-copyable by the Tensor Data Mover and B-operand
// fragments are K-contiguous in LDS.
// ---------------------------------------------------------------------------
__global__ __launch_bounds__(256)
void convert_copy_kernel(const float* __restrict__ in, __bf16* __restrict__ out)
{
  const size_t idx = (size_t)blockIdx.x * 256 + threadIdx.x;   // one float4 each
  float4 v = ((const float4*)in)[idx];
  v4bf o;
  o[0] = (__bf16)v.x; o[1] = (__bf16)v.y; o[2] = (__bf16)v.z; o[3] = (__bf16)v.w;
  *(v4bf*)(out + 4 * idx) = o;
}

__global__ __launch_bounds__(256)
void convert_transpose_kernel(const float* __restrict__ in /*[K][N]*/,
                              __bf16* __restrict__ out /*[N][K]*/, int K, int N)
{
  __shared__ float tile[32][33];
  const int tx = threadIdx.x & 31;          // 32 x 8 logical layout
  const int ty = threadIdx.x >> 5;
  const int bx = blockIdx.x * 32;           // N direction
  const int by = blockIdx.y * 32;           // K direction
  #pragma unroll
  for (int r = 0; r < 4; ++r)
    tile[ty + 8 * r][tx] = in[(size_t)(by + ty + 8 * r) * N + bx + tx];
  __syncthreads();
  #pragma unroll
  for (int r = 0; r < 4; ++r)
    out[(size_t)(bx + ty + 8 * r) * K + by + tx] = (__bf16)tile[tx][ty + 8 * r];
}

// ---------------------------------------------------------------------------
// TDM descriptor build + issue: 2D tile (tile_d0 x tile_d1) of 2-byte elems,
// row stride d0_stride (elems), global tile start gptr -> LDS at lds_addr.
// D# layout per CDNA5 ISA.
// ---------------------------------------------------------------------------
#if HAVE_TDM
__device__ __forceinline__ void tdm_load_2d(unsigned lds_addr, const __bf16* gptr,
                                            unsigned tensor_d0, unsigned tensor_d1,
                                            unsigned tile_d0, unsigned tile_d1,
                                            unsigned long long d0_stride)
{
  const unsigned long long ga = (unsigned long long)gptr;
  u32x4 g0;
  g0[0] = 1u;                                       // count=1 (valid), user mode
  g0[1] = lds_addr;                                 // lds_addr [63:32]
  g0[2] = (unsigned)(ga & 0xFFFFFFFFu);             // global_addr [95:64]
  g0[3] = (unsigned)((ga >> 32) & 0x1FFFFFFu)       // global_addr [120:96]
        | (2u << 30);                               // type=2 ("image")
  i32x8 g1;
  g1[0] = (int)(1u << 16);                          // workgroup_mask=0, data_size=1 (2B)
  g1[1] = (int)((tensor_d0 & 0xFFFFu) << 16);       // tensor_dim0[15:0] @ bits63:48
  g1[2] = (int)(((tensor_d0 >> 16) & 0xFFFFu) | ((tensor_d1 & 0xFFFFu) << 16));
  g1[3] = (int)(((tensor_d1 >> 16) & 0xFFFFu) | ((tile_d0 & 0xFFFFu) << 16));
  g1[4] = (int)(tile_d1 & 0xFFFFu);                 // tile_dim1; tile_dim2=0 (2D)
  g1[5] = (int)(d0_stride & 0xFFFFFFFFu);           // dim0_stride[31:0]
  g1[6] = (int)((d0_stride >> 32) & 0xFFFFu);       // dim0_stride[47:32]
  g1[7] = 0;
  const i32x4 gz = {0, 0, 0, 0};
#if __clang_major__ >= 23
  const i32x8 gz8 = {0, 0, 0, 0, 0, 0, 0, 0};
  __builtin_amdgcn_tensor_load_to_lds(g0, g1, gz, gz, gz8, 0);
#else
  __builtin_amdgcn_tensor_load_to_lds(g0, g1, gz, gz, 0);
#endif
}
#endif

__device__ __forceinline__ unsigned lds_off(const void* p)
{ return (unsigned)(unsigned long long)p; }   // low 32 bits of flat addr == LDS offset

// ---------------------------------------------------------------------------
// GEMM: C_bf16[M,N] = act(A_bf16[M,K] @ Bt_bf16[N,K]^T + bias).
// BM=128 BN=64 BK=32, 256 threads = 8 waves; wave -> 32x32 patch via 2x2
// v_wmma_f32_16x16x32_bf16. Tiles staged by TDM (double buffered, wave 0
// issues, TENSORcnt sync) into contiguous LDS; fragments are b128 LDS reads.
// ---------------------------------------------------------------------------
template<bool RELU>
__global__ __launch_bounds__(256)
void gemm_tdm(const __bf16* __restrict__ A, const __bf16* __restrict__ Bt,
              const float* __restrict__ bias, __bf16* __restrict__ C,
              int M, int Ncols, int K)
{
  __shared__ __attribute__((aligned(16))) __bf16 sA[2][128 * 32];  // row-major tile
  __shared__ __attribute__((aligned(16))) __bf16 sB[2][64 * 32];   // [n][k] tile

  const int t    = threadIdx.x;
  const int lane = t & 31;
  const int wid  = t >> 5;
  const int wm   = wid & 3;
  const int wn   = wid >> 2;
  const int bm   = blockIdx.y * 128;
  const int bn   = blockIdx.x * 64;
  const int m16  = lane & 15;
  const int kh   = lane >> 4;

  v8f acc[2][2] = {};

  auto stage = [&](int buf, int k0) {
#if HAVE_TDM
    if (wid == 0) {
      tdm_load_2d(lds_off(&sA[buf][0]), A  + (size_t)bm * K + k0,
                  (unsigned)K, (unsigned)M,     32u, 128u, (unsigned long long)K);
      tdm_load_2d(lds_off(&sB[buf][0]), Bt + (size_t)bn * K + k0,
                  (unsigned)K, (unsigned)Ncols, 32u, 64u,  (unsigned long long)K);
    }
#else
    { // cooperative fallback: same LDS layout
      const int ar = t >> 1, ac = (t & 1) * 16;
      const v8bf* sa = (const v8bf*)(A + (size_t)(bm + ar) * K + k0 + ac);
      *(v8bf*)&sA[buf][ar * 32 + ac]     = sa[0];
      *(v8bf*)&sA[buf][ar * 32 + ac + 8] = sa[1];
      if (t < 128) {
        const int br = t >> 1, bc = (t & 1) * 16;
        const v8bf* sb = (const v8bf*)(Bt + (size_t)(bn + br) * K + k0 + bc);
        *(v8bf*)&sB[buf][br * 32 + bc]     = sb[0];
        *(v8bf*)&sB[buf][br * 32 + bc + 8] = sb[1];
      }
    }
#endif
  };

  const int nk = K >> 5;
  stage(0, 0);
  for (int kt = 0; kt < nk; ++kt) {
    const int cur = kt & 1;
    if (kt + 1 < nk) {
      stage(cur ^ 1, (kt + 1) << 5);
#if HAVE_TDM
      if (wid == 0) __builtin_amdgcn_s_wait_tensorcnt(2);  // current tile done
#endif
    } else {
#if HAVE_TDM
      if (wid == 0) __builtin_amdgcn_s_wait_tensorcnt(0);
#endif
    }
    __syncthreads();

    v16bf afrag[2], bfrag[2];
    #pragma unroll
    for (int i = 0; i < 2; ++i) {
      // A 16x32 lane layout: lanes0-15 K={0..7,16..23}, lanes16-31 K={8..15,24..31}
      const __bf16* ar = &sA[cur][(wm * 32 + i * 16 + m16) * 32];
      v8bf lo = *(const v8bf*)(ar + kh * 8);
      v8bf hi = *(const v8bf*)(ar + 16 + kh * 8);
      afrag[i] = __builtin_shufflevector(lo, hi, 0,1,2,3,4,5,6,7,8,9,10,11,12,13,14,15);
    }
    #pragma unroll
    for (int j = 0; j < 2; ++j) {
      // B 32x16 lane layout: lane n=l&15, lanes0-15 K={0..7,16..23}, lanes16-31 K={8..15,24..31}
      const __bf16* br = &sB[cur][(wn * 32 + j * 16 + m16) * 32];
      v8bf lo = *(const v8bf*)(br + kh * 8);
      v8bf hi = *(const v8bf*)(br + 16 + kh * 8);
      bfrag[j] = __builtin_shufflevector(lo, hi, 0,1,2,3,4,5,6,7,8,9,10,11,12,13,14,15);
    }
    #pragma unroll
    for (int i = 0; i < 2; ++i)
      #pragma unroll
      for (int j = 0; j < 2; ++j)
        acc[i][j] = __builtin_amdgcn_wmma_f32_16x16x32_bf16(
            false, afrag[i], false, bfrag[j], (short)0, acc[i][j], false, false);
    __syncthreads();
  }

  // Epilogue: C layout lane l: N=l&15, M=r+8*(l>>4)
  #pragma unroll
  for (int i = 0; i < 2; ++i) {
    #pragma unroll
    for (int j = 0; j < 2; ++j) {
      const int gcol = bn + wn * 32 + j * 16 + m16;
      const float bv = bias[gcol];
      #pragma unroll
      for (int r = 0; r < 8; ++r) {
        const int grow = bm + wm * 32 + i * 16 + kh * 8 + r;
        float v = acc[i][j][r] + bv;
        if (RELU) v = fmaxf(v, 0.0f);
        C[(size_t)grow * Ncols + gcol] = (__bf16)v;
      }
    }
  }
}

// ---------------------------------------------------------------------------
// z = h2 @ W3 + b3 (K=256, 10 classes padded to 16 cols) + row norms, f32.
// ---------------------------------------------------------------------------
__global__ __launch_bounds__(256)
void zlayer_kernel(const __bf16* __restrict__ h2, const float* __restrict__ W3,
                   const float* __restrict__ b3, float* __restrict__ z,
                   float* __restrict__ norm)
{
  __shared__ float sW[DH2][NCLS];
  __shared__ float sb[NCLS];
  const int t = threadIdx.x;
  for (int idx = t; idx < DH2 * NCLS; idx += 256) sW[idx / NCLS][idx % NCLS] = W3[idx];
  if (t < NCLS) sb[t] = b3[t];
  __syncthreads();

  const int i = blockIdx.x * 256 + t;
  float zc[NCLS];
  #pragma unroll
  for (int c = 0; c < NCLS; ++c) zc[c] = sb[c];
  const __bf16* row = h2 + (size_t)i * DH2;
  for (int d = 0; d < DH2; ++d) {
    const float hv = (float)row[d];
    #pragma unroll
    for (int c = 0; c < NCLS; ++c) zc[c] = fmaf(hv, sW[d][c], zc[c]);
  }
  float nrm = 0.0f;
  #pragma unroll
  for (int c = 0; c < NCLS; ++c) { z[(size_t)i * 16 + c] = zc[c]; nrm = fmaf(zc[c], zc[c], nrm); }
  #pragma unroll
  for (int c = NCLS; c < 16; ++c) z[(size_t)i * 16 + c] = 0.0f;
  norm[i] = nrm;
}

// ---------------------------------------------------------------------------
// Fused Gram + exp kernel + per-class aggregation; deterministic partials.
// ---------------------------------------------------------------------------
__global__ __launch_bounds__(256)
void pairwise_kernel(const float* __restrict__ z, const float* __restrict__ norm,
                     const int* __restrict__ y, float* __restrict__ partial)
{
  __shared__ float sZ[128][NCLS];
  __shared__ float sN[128];
  __shared__ int   sY[128];
  const int t  = threadIdx.x;
  const int jc = blockIdx.x;
  const int i  = blockIdx.y * 256 + t;

  float zi[NCLS], acc[NCLS];
  #pragma unroll
  for (int c = 0; c < NCLS; ++c) { zi[c] = z[(size_t)i * 16 + c]; acc[c] = 0.0f; }
  const float ni = norm[i];
  const int j0 = jc * JCHUNK;

  for (int jt = 0; jt < JCHUNK; jt += 128) {
    __syncthreads();
    for (int idx = t; idx < 128 * NCLS; idx += 256) {
      const int jj = idx / NCLS, c = idx % NCLS;
      sZ[jj][c] = z[(size_t)(j0 + jt + jj) * 16 + c];
    }
    if (t < 128) { sN[t] = norm[j0 + jt + t]; sY[t] = y[j0 + jt + t]; }
    if (jt + 128 < JCHUNK)   // pull next tile toward the caches (global_prefetch_b8)
      __builtin_prefetch(z + (size_t)(j0 + jt + 128 + t) * 16, 0, 1);
    __syncthreads();

    #pragma unroll 4
    for (int jj = 0; jj < 128; ++jj) {
      float dot = 0.0f;
      #pragma unroll
      for (int c = 0; c < NCLS; ++c) dot = fmaf(zi[c], sZ[jj][c], dot);
      const float dist = ni + sN[jj] - 2.0f * dot;
      float e = __expf(-0.5f * GAMMA * dist);
      if (j0 + jt + jj == i) e = 0.0f;          // zero diagonal
      const int yj = sY[jj];
      #pragma unroll
      for (int c = 0; c < NCLS; ++c) acc[c] += (yj == c) ? e : 0.0f;
    }
  }
  float* out = partial + ((size_t)jc * NROWS + i) * NCLS;
  #pragma unroll
  for (int c = 0; c < NCLS; ++c) out[c] = acc[c];
}

// ---------------------------------------------------------------------------
// class_dists = EPS + sum_jc partial; probs = log(cd/sum); per-row NLL.
// ---------------------------------------------------------------------------
__global__ __launch_bounds__(256)
void probs_kernel(const float* __restrict__ partial, const int* __restrict__ y,
                  float* __restrict__ probs, float* __restrict__ nll)
{
  const int i = blockIdx.x * 256 + threadIdx.x;
  float cd[NCLS];
  #pragma unroll
  for (int c = 0; c < NCLS; ++c) cd[c] = KEPS;
  for (int jc = 0; jc < NJC; ++jc) {
    const float* p = partial + ((size_t)jc * NROWS + i) * NCLS;
    #pragma unroll
    for (int c = 0; c < NCLS; ++c) cd[c] += p[c];
  }
  float s = 0.0f;
  #pragma unroll
  for (int c = 0; c < NCLS; ++c) s += cd[c];
  const float ls = logf(s);
  const int yi = y[i];
  float sel = 0.0f;
  #pragma unroll
  for (int c = 0; c < NCLS; ++c) {
    const float pc = logf(cd[c]) - ls;
    probs[(size_t)i * NCLS + c] = pc;
    sel += (c == yi) ? pc : 0.0f;
  }
  nll[i] = -sel;
}

// ---------------------------------------------------------------------------
// Deterministic single-block loss reduction; writes the two scalar outputs.
// ---------------------------------------------------------------------------
__global__ __launch_bounds__(256)
void loss_reduce_kernel(const float* __restrict__ nll, float* __restrict__ out)
{
  __shared__ float red[256];
  const int t = threadIdx.x;
  float s = 0.0f;
  for (int i = t; i < NROWS; i += 256) s += nll[i];
  red[t] = s;
  __syncthreads();
  for (int off = 128; off > 0; off >>= 1) {
    if (t < off) red[t] += red[t + off];
    __syncthreads();
  }
  if (t == 0) {
    const float total = red[0];
    out[NROWS * NCLS + 0] = total / (float)NROWS;
    out[NROWS * NCLS + 1] = total;
  }
}

extern "C" void kernel_launch(void* const* d_in, const int* in_sizes, int n_in,
                              void* d_out, int out_size, void* d_ws, size_t ws_size,
                              hipStream_t stream)
{
  const float* x  = (const float*)d_in[0];
  const int*   y  = (const int*)  d_in[1];
  const float* W1 = (const float*)d_in[2];
  const float* b1 = (const float*)d_in[3];
  const float* W2 = (const float*)d_in[4];
  const float* b2 = (const float*)d_in[5];
  const float* W3 = (const float*)d_in[6];
  const float* b3 = (const float*)d_in[7];
  float* out = (float*)d_out;

  char* ws = (char*)d_ws;
  size_t off = 0;
  auto take = [&](size_t bytes) -> char* {
    char* p = ws + off;
    off = (off + bytes + 255) & ~(size_t)255;
    return p;
  };
  __bf16* xbf  = (__bf16*)take((size_t)NROWS * XDIM * sizeof(__bf16));  // 16 MB
  __bf16* W1t  = (__bf16*)take((size_t)DH1 * XDIM * sizeof(__bf16));    // 1 MB
  __bf16* W2t  = (__bf16*)take((size_t)DH2 * DH1 * sizeof(__bf16));     // 256 KB
  __bf16* h    = (__bf16*)take((size_t)NROWS * DH1 * sizeof(__bf16));   // 8 MB
  __bf16* h2   = (__bf16*)take((size_t)NROWS * DH2 * sizeof(__bf16));   // 4 MB
  float*  z    = (float*) take((size_t)NROWS * 16 * sizeof(float));     // 512 KB
  float*  norm = (float*) take((size_t)NROWS * sizeof(float));          // 32 KB
  float*  part = (float*) take((size_t)NJC * NROWS * NCLS * sizeof(float)); // 5.25 MB
  float*  nll  = (float*) take((size_t)NROWS * sizeof(float));          // 32 KB
  (void)in_sizes; (void)n_in; (void)out_size; (void)ws_size;

  convert_copy_kernel<<<(NROWS * XDIM / 4) / 256, 256, 0, stream>>>(x, xbf);
  convert_transpose_kernel<<<dim3(DH1 / 32, XDIM / 32), 256, 0, stream>>>(W1, W1t, XDIM, DH1);
  convert_transpose_kernel<<<dim3(DH2 / 32, DH1 / 32), 256, 0, stream>>>(W2, W2t, DH1, DH2);

  gemm_tdm<true ><<<dim3(DH1 / 64, NROWS / 128), 256, 0, stream>>>(xbf, W1t, b1, h,  NROWS, DH1, XDIM);
  gemm_tdm<false><<<dim3(DH2 / 64, NROWS / 128), 256, 0, stream>>>(h,   W2t, b2, h2, NROWS, DH2, DH1);

  zlayer_kernel   <<<NROWS / 256, 256, 0, stream>>>(h2, W3, b3, z, norm);
  pairwise_kernel <<<dim3(NJC, NROWS / 256), 256, 0, stream>>>(z, norm, y, part);
  probs_kernel    <<<NROWS / 256, 256, 0, stream>>>(part, y, out, nll);
  loss_reduce_kernel<<<1, 256, 0, stream>>>(nll, out);
}
